// DSMoE_53386443489942
// MI455X (gfx1250) — matrix operand order repaired and checked by reading, not compiled, not run on hardware
//
#include <hip/hip_runtime.h>
#include <hip/hip_bf16.h>
#include <stdint.h>

// Problem constants (match reference)
#define BB 4
#define SS 1024
#define HH 1024
#define II 2048
#define EE 8
#define KK 2
#define TT (BB * SS)           // 4096 tokens
#define MAXROWS (TT * KK)      // 8192 routed rows total

#define GK 64                  // K-tile depth
#define LSTR 72                // LDS row stride in halfwords (144B = 9*16, 16B aligned rows)

typedef __attribute__((ext_vector_type(16))) __bf16 v16bf;
typedef __attribute__((ext_vector_type(8)))  float  v8f;

union FragBF { v16bf v; unsigned u[8]; uint4 q[2]; };

static __device__ __forceinline__ unsigned short f32_to_bf16(float f) {
    unsigned u = __float_as_uint(f);
    unsigned r = 0x7FFFu + ((u >> 16) & 1u);   // round-to-nearest-even
    u += r;
    return (unsigned short)(u >> 16);
}

// ---- CDNA5 async memory->LDS copy (ISA §10 / §15.18, ASYNCcnt-tracked) ----
// VDST = LDS byte address (wave-relative; low 32 bits of flat shared ptr),
// VADDR = 64-bit global address, no SADDR ("off").
static __device__ __forceinline__ void async_load_b128(const void* gptr, void* lptr) {
    unsigned lds = (unsigned)(uintptr_t)lptr;
    unsigned long long ga = (unsigned long long)(uintptr_t)gptr;
    asm volatile("global_load_async_to_lds_b128 %0, %1, off" :: "v"(lds), "v"(ga) : "memory");
}
static __device__ __forceinline__ void wait_async0() {
    asm volatile("s_wait_asynccnt 0" ::: "memory");
}

// ---------------------------------------------------------------- init
__global__ void moe_init_kernel(int* cnt) {
    int t = threadIdx.x;
    if (t < EE) cnt[t] = 0;
}

// ---------------------------------------------------------------- x -> bf16 (row-major kept)
__global__ __launch_bounds__(256) void moe_cvt_kernel(const float* __restrict__ x,
                                                      unsigned short* __restrict__ xb) {
    int gid = blockIdx.x * 256 + threadIdx.x;
    int i = gid * 4;
    const float4 v = *(const float4*)(x + i);
    ushort4 o;
    o.x = f32_to_bf16(v.x); o.y = f32_to_bf16(v.y);
    o.z = f32_to_bf16(v.z); o.w = f32_to_bf16(v.w);
    *(ushort4*)(xb + i) = o;
}

// ---------------------------------------------------------------- transpose + cvt: src f32 [R][C] -> dst bf16 [C][R]
__global__ __launch_bounds__(256) void moe_tcvt_kernel(const float* __restrict__ src,
                                                       unsigned short* __restrict__ dst,
                                                       int R, int C) {
    const size_t slice = (size_t)R * C * blockIdx.z;
    src += slice; dst += slice;
    __shared__ float tile[32][33];
    const int r0 = blockIdx.y * 32, c0 = blockIdx.x * 32;
    const int tr = threadIdx.x >> 3;        // 0..31
    const int tc = (threadIdx.x & 7) * 4;   // 0,4,..,28

    const float4 v = *(const float4*)(src + (size_t)(r0 + tr) * C + c0 + tc);
    tile[tr][tc + 0] = v.x; tile[tr][tc + 1] = v.y;
    tile[tr][tc + 2] = v.z; tile[tr][tc + 3] = v.w;
    __syncthreads();

    ushort4 o;
    o.x = f32_to_bf16(tile[tc + 0][tr]);
    o.y = f32_to_bf16(tile[tc + 1][tr]);
    o.z = f32_to_bf16(tile[tc + 2][tr]);
    o.w = f32_to_bf16(tile[tc + 3][tr]);
    *(ushort4*)(dst + (size_t)(c0 + tr) * R + r0 + tc) = o;
}

// ---------------------------------------------------------------- router: one wave32 per token
__global__ __launch_bounds__(256) void moe_router_kernel(
    const float* __restrict__ x, const float* __restrict__ gate_w,
    const float* __restrict__ ebias,
    int* __restrict__ sel_idx, float* __restrict__ sel_w, int* __restrict__ cnt) {
    int gid  = blockIdx.x * 256 + threadIdx.x;
    int tok  = gid >> 5;
    int lane = gid & 31;

    float acc[EE];
#pragma unroll
    for (int e = 0; e < EE; ++e) acc[e] = 0.0f;

    for (int h = lane; h < HH; h += 32) {
        float xv = x[(size_t)tok * HH + h];
#pragma unroll
        for (int e = 0; e < EE; ++e) acc[e] += xv * gate_w[(size_t)e * HH + h];
    }
#pragma unroll
    for (int off = 16; off > 0; off >>= 1) {
#pragma unroll
        for (int e = 0; e < EE; ++e) acc[e] += __shfl_xor(acc[e], off, 32);
    }
    if (lane == 0) {
        float b[EE];
#pragma unroll
        for (int e = 0; e < EE; ++e) b[e] = acc[e] + ebias[e];
        int i1 = 0;
#pragma unroll
        for (int e = 1; e < EE; ++e) if (b[e] > b[i1]) i1 = e;
        int i2 = (i1 == 0) ? 1 : 0;
#pragma unroll
        for (int e = 0; e < EE; ++e) if (e != i1 && b[e] > b[i2]) i2 = e;
        int j1 = 0;
#pragma unroll
        for (int e = 1; e < EE; ++e) if (acc[e] > acc[j1]) j1 = e;
        int j2 = (j1 == 0) ? 1 : 0;
#pragma unroll
        for (int e = 0; e < EE; ++e) if (e != j1 && acc[e] > acc[j2]) j2 = e;
        float w0 = 1.0f / (1.0f + __expf(-acc[j1]));
        float w1 = 1.0f / (1.0f + __expf(-acc[j2]));
        float s = w0 + w1;
        w0 /= s; w1 /= s;
        sel_idx[tok * 2 + 0] = i1;
        sel_idx[tok * 2 + 1] = i2;
        sel_w[tok * 2 + 0] = w0;
        sel_w[tok * 2 + 1] = w1;
        atomicAdd(&cnt[i1], 1);
        atomicAdd(&cnt[i2], 1);
    }
}

// ---------------------------------------------------------------- prefix + viol
__global__ void moe_prefix_kernel(const int* __restrict__ cnt, int* __restrict__ eoff,
                                  int* __restrict__ cursor, float* __restrict__ out_viol) {
    if (threadIdx.x == 0 && blockIdx.x == 0) {
        int run = 0;
        float mx = 0.0f;
        const float total = (float)(TT * KK);
        for (int e = 0; e < EE; ++e) {
            eoff[e] = run;
            cursor[e] = run;
            run += cnt[e];
            float f = (float)cnt[e] / total;
            mx = fmaxf(mx, f);
        }
        const float mean = 1.0f / (float)EE;
        *out_viol = (mx - mean) / mean;
    }
}

// ---------------------------------------------------------------- scatter compaction
__global__ __launch_bounds__(256) void moe_scatter_kernel(
    const int* __restrict__ sel_idx, const float* __restrict__ sel_w,
    int* __restrict__ cursor, int* __restrict__ tok_ids, float* __restrict__ cwc) {
    int t = blockIdx.x * 256 + threadIdx.x;
#pragma unroll
    for (int j = 0; j < KK; ++j) {
        int e = sel_idx[t * 2 + j];
        int p = atomicAdd(&cursor[e], 1);
        tok_ids[p] = t;
        cwc[p] = sel_w[t * 2 + j];
    }
}

// ---------------------------------------------------------------- GEMM1: hidden = silu(X@Wg)*(X@Wu)
// Weights pre-transposed bf16: WT [*][N=I][K=H] (K contiguous)
__global__ __launch_bounds__(256) void moe_gemm1_kernel(
    const unsigned short* __restrict__ xb,
    const unsigned short* __restrict__ WgT_all, const unsigned short* __restrict__ WuT_all,
    unsigned short* __restrict__ hidden,
    const int* __restrict__ tok_ids, const int* __restrict__ eoff,
    const int* __restrict__ cnt, int shared_mode) {
    const int e = blockIdx.z;
    int Ne, base;
    const unsigned short *WgT, *WuT;
    if (shared_mode) {
        Ne = TT; base = 0; WgT = WgT_all; WuT = WuT_all;
    } else {
        Ne = cnt[e]; base = eoff[e];
        size_t wo = (size_t)e * II * HH;
        WgT = WgT_all + wo; WuT = WuT_all + wo;
    }
    const int m0 = blockIdx.y * 64;
    if (m0 >= Ne) return;
    const int n0 = blockIdx.x * 64;

    __shared__ int toks[64];
    __shared__ __align__(16) unsigned short As[2][64 * LSTR];
    __shared__ __align__(16) unsigned short Bgs[2][64 * LSTR];
    __shared__ __align__(16) unsigned short Bus[2][64 * LSTR];

    const int t = threadIdx.x;
    if (t < 64) {
        int r = m0 + t;
        toks[t] = (r < Ne) ? (shared_mode ? r : tok_ids[base + r]) : -1;
    }
    __syncthreads();

    const int wid = t >> 5, lane = t & 31;
    const int wm = wid & 1, wn = wid >> 1;       // 2 x 4 waves over 64x64 tile
    const int hs = lane >> 4, ll = lane & 15;
    v8f cg[2] = {};
    v8f cu[2] = {};

    // fill indices: 64 rows x 64 K halfwords, 32B (16 halfwords) per thread
    const int fr = t >> 2, fc = (t & 3) * 16;
    const int tokr = toks[fr];

    // invalid padded A rows: zero both buffers once (async never writes them)
    if (tokr < 0) {
        const uint4 z = {0, 0, 0, 0};
        *(uint4*)&As[0][fr * LSTR + fc]     = z;
        *(uint4*)&As[0][fr * LSTR + fc + 8] = z;
        *(uint4*)&As[1][fr * LSTR + fc]     = z;
        *(uint4*)&As[1][fr * LSTR + fc + 8] = z;
    }

    // ---- prologue: async stage tile 0 into buf 0
    if (tokr >= 0) {
        const unsigned short* g = xb + (size_t)tokr * HH + fc;
        async_load_b128(g,     &As[0][fr * LSTR + fc]);
        async_load_b128(g + 8, &As[0][fr * LSTR + fc + 8]);
    }
    {
        const unsigned short* pg = WgT + (size_t)(n0 + fr) * HH + fc;
        const unsigned short* pu = WuT + (size_t)(n0 + fr) * HH + fc;
        async_load_b128(pg,     &Bgs[0][fr * LSTR + fc]);
        async_load_b128(pg + 8, &Bgs[0][fr * LSTR + fc + 8]);
        async_load_b128(pu,     &Bus[0][fr * LSTR + fc]);
        async_load_b128(pu + 8, &Bus[0][fr * LSTR + fc + 8]);
    }
    wait_async0();
    __syncthreads();

    const int NKT = HH / GK;
    for (int kt = 0; kt < NKT; ++kt) {
        const int cur = kt & 1;
        const bool has_next = (kt + 1) < NKT;
        // ---- async stage next tile into the other buffer (overlaps WMMA below)
        if (has_next) {
            const int nb = cur ^ 1;
            const int k0n = (kt + 1) * GK;
            if (tokr >= 0) {
                const unsigned short* g = xb + (size_t)tokr * HH + k0n + fc;
                async_load_b128(g,     &As[nb][fr * LSTR + fc]);
                async_load_b128(g + 8, &As[nb][fr * LSTR + fc + 8]);
            }
            const unsigned short* pg = WgT + (size_t)(n0 + fr) * HH + k0n + fc;
            const unsigned short* pu = WuT + (size_t)(n0 + fr) * HH + k0n + fc;
            async_load_b128(pg,     &Bgs[nb][fr * LSTR + fc]);
            async_load_b128(pg + 8, &Bgs[nb][fr * LSTR + fc + 8]);
            async_load_b128(pu,     &Bus[nb][fr * LSTR + fc]);
            async_load_b128(pu + 8, &Bus[nb][fr * LSTR + fc + 8]);
        }
        // ---- compute on current LDS tile: 2 K-steps x 2 M-frags x {gate,up} = 8 WMMA
#pragma unroll
        for (int ks = 0; ks < GK; ks += 32) {
            FragBF a[2], bg, bu;
#pragma unroll
            for (int mi = 0; mi < 2; ++mi) {
                const int M = wm * 32 + mi * 16 + ll;
                a[mi].q[0] = *(const uint4*)&As[cur][M * LSTR + ks + hs * 16];
                a[mi].q[1] = *(const uint4*)&As[cur][M * LSTR + ks + hs * 16 + 8];
            }
            const int N = wn * 16 + ll;
            bg.q[0] = *(const uint4*)&Bgs[cur][N * LSTR + ks + hs * 16];
            bg.q[1] = *(const uint4*)&Bgs[cur][N * LSTR + ks + hs * 16 + 8];
            bu.q[0] = *(const uint4*)&Bus[cur][N * LSTR + ks + hs * 16];
            bu.q[1] = *(const uint4*)&Bus[cur][N * LSTR + ks + hs * 16 + 8];
#pragma unroll
            for (int mi = 0; mi < 2; ++mi) {
                cg[mi] = __builtin_amdgcn_wmma_f32_16x16x32_bf16(false, a[mi].v, false, bg.v,
                                                                 (short)0, cg[mi], false, false);
                cu[mi] = __builtin_amdgcn_wmma_f32_16x16x32_bf16(false, a[mi].v, false, bu.v,
                                                                 (short)0, cu[mi], false, false);
            }
        }
        if (has_next) wait_async0();
        __syncthreads();
    }

    // ---- SwiGLU epilogue -> hidden (bf16)
#pragma unroll
    for (int mi = 0; mi < 2; ++mi) {
#pragma unroll
        for (int v = 0; v < 8; ++v) {
            int M = wm * 32 + mi * 16 + v + 8 * hs;
            int row = m0 + M;
            if (row < Ne) {
                float g = cg[mi][v];
                float u = cu[mi][v];
                float h = (g / (1.0f + __expf(-g))) * u;
                int col = n0 + wn * 16 + ll;
                hidden[(size_t)(base + row) * II + col] = f32_to_bf16(h);
            }
        }
    }
}

// ---------------------------------------------------------------- GEMM2: out (+)= cw * (hidden @ Wd)
// WdT [*][N=H][K=I] (K contiguous)
__global__ __launch_bounds__(256) void moe_gemm2_kernel(
    const unsigned short* __restrict__ hidden, const unsigned short* __restrict__ WdT_all,
    float* __restrict__ out,
    const int* __restrict__ tok_ids, const float* __restrict__ cwc,
    const int* __restrict__ eoff, const int* __restrict__ cnt, int shared_mode) {
    const int e = blockIdx.z;
    int Ne, base;
    const unsigned short* WdT;
    if (shared_mode) {
        Ne = TT; base = 0; WdT = WdT_all;
    } else {
        Ne = cnt[e]; base = eoff[e];
        WdT = WdT_all + (size_t)e * HH * II;
    }
    const int m0 = blockIdx.y * 64;
    if (m0 >= Ne) return;
    const int n0 = blockIdx.x * 64;

    __shared__ int toks[64];
    __shared__ float cws[64];
    __shared__ __align__(16) unsigned short As[2][64 * LSTR];
    __shared__ __align__(16) unsigned short Bds[2][64 * LSTR];

    const int t = threadIdx.x;
    if (t < 64) {
        int r = m0 + t;
        if (r < Ne) {
            toks[t] = shared_mode ? r : tok_ids[base + r];
            cws[t]  = shared_mode ? 1.0f : cwc[base + r];
        } else {
            toks[t] = -1;
            cws[t]  = 0.0f;
        }
    }
    __syncthreads();

    const int wid = t >> 5, lane = t & 31;
    const int wm = wid & 1, wn = wid >> 1;
    const int hs = lane >> 4, ll = lane & 15;
    v8f c[2] = {};

    const int fr = t >> 2, fc = (t & 3) * 16;
    const int rowr = m0 + fr;
    const bool rvalid = rowr < Ne;

    if (!rvalid) {
        const uint4 z = {0, 0, 0, 0};
        *(uint4*)&As[0][fr * LSTR + fc]     = z;
        *(uint4*)&As[0][fr * LSTR + fc + 8] = z;
        *(uint4*)&As[1][fr * LSTR + fc]     = z;
        *(uint4*)&As[1][fr * LSTR + fc + 8] = z;
    }

    if (rvalid) {
        const unsigned short* g = hidden + (size_t)(base + rowr) * II + fc;
        async_load_b128(g,     &As[0][fr * LSTR + fc]);
        async_load_b128(g + 8, &As[0][fr * LSTR + fc + 8]);
    }
    {
        const unsigned short* pd = WdT + (size_t)(n0 + fr) * II + fc;
        async_load_b128(pd,     &Bds[0][fr * LSTR + fc]);
        async_load_b128(pd + 8, &Bds[0][fr * LSTR + fc + 8]);
    }
    wait_async0();
    __syncthreads();

    const int NKT = II / GK;
    for (int kt = 0; kt < NKT; ++kt) {
        const int cur = kt & 1;
        const bool has_next = (kt + 1) < NKT;
        if (has_next) {
            const int nb = cur ^ 1;
            const int k0n = (kt + 1) * GK;
            if (rvalid) {
                const unsigned short* g = hidden + (size_t)(base + rowr) * II + k0n + fc;
                async_load_b128(g,     &As[nb][fr * LSTR + fc]);
                async_load_b128(g + 8, &As[nb][fr * LSTR + fc + 8]);
            }
            const unsigned short* pd = WdT + (size_t)(n0 + fr) * II + k0n + fc;
            async_load_b128(pd,     &Bds[nb][fr * LSTR + fc]);
            async_load_b128(pd + 8, &Bds[nb][fr * LSTR + fc + 8]);
        }
#pragma unroll
        for (int ks = 0; ks < GK; ks += 32) {
            FragBF a[2], bd;
#pragma unroll
            for (int mi = 0; mi < 2; ++mi) {
                const int M = wm * 32 + mi * 16 + ll;
                a[mi].q[0] = *(const uint4*)&As[cur][M * LSTR + ks + hs * 16];
                a[mi].q[1] = *(const uint4*)&As[cur][M * LSTR + ks + hs * 16 + 8];
            }
            const int N = wn * 16 + ll;
            bd.q[0] = *(const uint4*)&Bds[cur][N * LSTR + ks + hs * 16];
            bd.q[1] = *(const uint4*)&Bds[cur][N * LSTR + ks + hs * 16 + 8];
#pragma unroll
            for (int mi = 0; mi < 2; ++mi) {
                c[mi] = __builtin_amdgcn_wmma_f32_16x16x32_bf16(false, a[mi].v, false, bd.v,
                                                                (short)0, c[mi], false, false);
            }
        }
        if (has_next) wait_async0();
        __syncthreads();
    }

#pragma unroll
    for (int mi = 0; mi < 2; ++mi) {
#pragma unroll
        for (int v = 0; v < 8; ++v) {
            int M = wm * 32 + mi * 16 + v + 8 * hs;
            int row = m0 + M;
            if (row < Ne) {
                int tok = toks[M];
                float val = c[mi][v] * cws[M];
                int col = n0 + wn * 16 + ll;
                if (shared_mode)
                    out[(size_t)tok * HH + col] = val;      // first writer: plain store
                else
                    atomicAdd(&out[(size_t)tok * HH + col], val);
            }
        }
    }
}

// ---------------------------------------------------------------- launch
extern "C" void kernel_launch(void* const* d_in, const int* in_sizes, int n_in,
                              void* d_out, int out_size, void* d_ws, size_t ws_size,
                              hipStream_t stream) {
    const float* x      = (const float*)d_in[0];
    const float* gate_w = (const float*)d_in[1];
    const float* ebias  = (const float*)d_in[2];
    const float* Wg     = (const float*)d_in[3];
    const float* Wu     = (const float*)d_in[4];
    const float* Wd     = (const float*)d_in[5];
    const float* Sg     = (const float*)d_in[6];
    const float* Su     = (const float*)d_in[7];
    const float* Sd     = (const float*)d_in[8];
    float* out = (float*)d_out;

    // workspace layout
    char* ws = (char*)d_ws;
    size_t o = 0;
    unsigned short* xb     = (unsigned short*)(ws + o); o += (size_t)TT * HH * 2;          // 8.39 MB
    unsigned short* hidden = (unsigned short*)(ws + o); o += (size_t)MAXROWS * II * 2;     // 33.55 MB
    unsigned short* WgT    = (unsigned short*)(ws + o); o += (size_t)EE * HH * II * 2;     // 33.55 MB
    unsigned short* WuT    = (unsigned short*)(ws + o); o += (size_t)EE * HH * II * 2;     // 33.55 MB
    unsigned short* WdT    = (unsigned short*)(ws + o); o += (size_t)EE * HH * II * 2;     // 33.55 MB
    unsigned short* SgT    = (unsigned short*)(ws + o); o += (size_t)HH * II * 2;          // 4.19 MB
    unsigned short* SuT    = (unsigned short*)(ws + o); o += (size_t)HH * II * 2;
    unsigned short* SdT    = (unsigned short*)(ws + o); o += (size_t)HH * II * 2;
    int*   tok_ids = (int*)  (ws + o); o += (size_t)MAXROWS * 4;
    float* cwc     = (float*)(ws + o); o += (size_t)MAXROWS * 4;
    int*   sel_idx = (int*)  (ws + o); o += (size_t)TT * KK * 4;
    float* sel_w   = (float*)(ws + o); o += (size_t)TT * KK * 4;
    int*   cnt     = (int*)  (ws + o); o += 256;
    int*   eoff    = (int*)  (ws + o); o += 256;
    int*   cursor  = (int*)  (ws + o); o += 256;
    (void)ws_size; (void)in_sizes; (void)n_in; (void)out_size;

    // 1. zero per-expert counters
    moe_init_kernel<<<1, 32, 0, stream>>>(cnt);
    // 2. x -> bf16
    moe_cvt_kernel<<<(TT * HH) / (256 * 4), 256, 0, stream>>>(x, xb);
    // 3. weights -> bf16, transposed to [N][K]
    moe_tcvt_kernel<<<dim3(II / 32, HH / 32, EE), 256, 0, stream>>>(Wg, WgT, HH, II);
    moe_tcvt_kernel<<<dim3(II / 32, HH / 32, EE), 256, 0, stream>>>(Wu, WuT, HH, II);
    moe_tcvt_kernel<<<dim3(HH / 32, II / 32, EE), 256, 0, stream>>>(Wd, WdT, II, HH);
    moe_tcvt_kernel<<<dim3(II / 32, HH / 32, 1 ), 256, 0, stream>>>(Sg, SgT, HH, II);
    moe_tcvt_kernel<<<dim3(II / 32, HH / 32, 1 ), 256, 0, stream>>>(Su, SuT, HH, II);
    moe_tcvt_kernel<<<dim3(HH / 32, II / 32, 1 ), 256, 0, stream>>>(Sd, SdT, II, HH);
    // 4. router
    moe_router_kernel<<<(TT * 32) / 256, 256, 0, stream>>>(x, gate_w, ebias, sel_idx, sel_w, cnt);
    // 5. prefix offsets + viol scalar (out[T*H])
    moe_prefix_kernel<<<1, 1, 0, stream>>>(cnt, eoff, cursor, out + (size_t)TT * HH);
    // 6. compact token lists per expert
    moe_scatter_kernel<<<TT / 256, 256, 0, stream>>>(sel_idx, sel_w, cursor, tok_ids, cwc);

    // 7. shared expert (all T tokens), plain-store epilogue
    moe_gemm1_kernel<<<dim3(II / 64, TT / 64, 1), 256, 0, stream>>>(
        xb, SgT, SuT, hidden, tok_ids, eoff, cnt, 1);
    moe_gemm2_kernel<<<dim3(HH / 64, TT / 64, 1), 256, 0, stream>>>(
        hidden, SdT, out, tok_ids, cwc, eoff, cnt, 1);
    // 8. routed experts (worst-case grid, device-side early exit), atomic-add epilogue
    moe_gemm1_kernel<<<dim3(II / 64, TT / 64, EE), 256, 0, stream>>>(
        xb, WgT, WuT, hidden, tok_ids, eoff, cnt, 0);
    moe_gemm2_kernel<<<dim3(HH / 64, TT / 64, EE), 256, 0, stream>>>(
        hidden, WdT, out, tok_ids, cwc, eoff, cnt, 0);
}
